// SparseMoE_48928267436600
// MI455X (gfx1250) — compile-verified, matrix operand order checked by reading, not run on hardware
//
#include <hip/hip_runtime.h>

// Problem dims (match reference)
#define B_TOK 4096
#define S_DIM 2048
#define P_DIM 720
#define E_NUM 16
#define K_TOP 4

typedef __attribute__((ext_vector_type(16))) _Float16 v16h;
typedef __attribute__((ext_vector_type(8)))  _Float16 v8h;
typedef __attribute__((ext_vector_type(8)))  float    v8f;

union HF16 { v16h v; v8h h[2]; };

// --- gfx1250 async global->LDS helpers -------------------------------------
__device__ __forceinline__ unsigned lds_off(const void* p) {
    // generic pointer to LDS: addr[31:0] is the wave-relative LDS byte offset
    return (unsigned)(unsigned long long)p;
}
__device__ __forceinline__ void async_ld_b128(const void* gptr, void* lptr) {
    asm volatile("global_load_async_to_lds_b128 %0, %1, off"
                 :: "v"(lds_off(lptr)), "v"((unsigned long long)gptr)
                 : "memory");
}
__device__ __forceinline__ void async_wait0() {
    asm volatile("s_wait_asynccnt 0x0" ::: "memory");
}

// ---------------------------------------------------------------------------
// Kernel 1: elementwise f32 -> f16 conversion. 8 elems / thread.
// ---------------------------------------------------------------------------
__launch_bounds__(256)
__global__ void convert_f32_to_f16(const float* __restrict__ src, _Float16* __restrict__ dst) {
    long i = ((long)blockIdx.x * 256 + threadIdx.x) * 8;
    const float4 f0 = *(const float4*)&src[i];
    const float4 f1 = *(const float4*)&src[i + 4];
    v8h h;
    h[0] = (_Float16)f0.x; h[1] = (_Float16)f0.y; h[2] = (_Float16)f0.z; h[3] = (_Float16)f0.w;
    h[4] = (_Float16)f1.x; h[5] = (_Float16)f1.y; h[6] = (_Float16)f1.z; h[7] = (_Float16)f1.w;
    *(v8h*)&dst[i] = h;
}

// ---------------------------------------------------------------------------
// Kernel 2: gating. One wave per token: 16 dot products of len 2048,
// wave reduction, top-4 + softmax.
// ---------------------------------------------------------------------------
__launch_bounds__(256)
__global__ void moe_gate_topk(const float* __restrict__ x,
                              const float* __restrict__ gate_w,
                              const float* __restrict__ gate_b,
                              int*   __restrict__ topk_idx,
                              float* __restrict__ topk_w) {
    const int token = blockIdx.x * 8 + (threadIdx.x >> 5);
    const int lane  = threadIdx.x & 31;

    float acc[E_NUM];
#pragma unroll
    for (int e = 0; e < E_NUM; ++e) acc[e] = 0.f;

    const float* xr = &x[(long)token * S_DIM];
    for (int s = lane; s < S_DIM; s += 32) {
        const float xv = xr[s];
#pragma unroll
        for (int e = 0; e < E_NUM; ++e) acc[e] += xv * gate_w[e * S_DIM + s];
    }
#pragma unroll
    for (int e = 0; e < E_NUM; ++e) {
#pragma unroll
        for (int off = 16; off >= 1; off >>= 1) acc[e] += __shfl_xor(acc[e], off, 32);
        acc[e] += gate_b[e];
    }

    unsigned used = 0;
    float vals[K_TOP]; int idxs[K_TOP];
#pragma unroll
    for (int k = 0; k < K_TOP; ++k) {
        float bv = -3.4e38f; int bi = 0;
#pragma unroll
        for (int e = 0; e < E_NUM; ++e) {
            const bool freee = !((used >> e) & 1u);
            if (freee && acc[e] > bv) { bv = acc[e]; bi = e; }
        }
        used |= 1u << bi; vals[k] = bv; idxs[k] = bi;
    }
    const float mx = vals[0];
    float ex[K_TOP], sum = 0.f;
#pragma unroll
    for (int k = 0; k < K_TOP; ++k) { ex[k] = __expf(vals[k] - mx); sum += ex[k]; }
    const float inv = 1.f / sum;
    if (lane == 0) {
#pragma unroll
        for (int k = 0; k < K_TOP; ++k) {
            topk_idx[token * K_TOP + k] = idxs[k];
            topk_w[token * K_TOP + k]   = ex[k] * inv;
        }
    }
}

// ---------------------------------------------------------------------------
// Kernel 3: deterministic per-expert compaction (ordered prefix-scan).
// ---------------------------------------------------------------------------
__launch_bounds__(256)
__global__ void moe_build_groups(const int*   __restrict__ topk_idx,
                                 const float* __restrict__ topk_w,
                                 int*   __restrict__ etok,
                                 float* __restrict__ ewt,
                                 int*   __restrict__ counts) {
    const int e = blockIdx.x;
    __shared__ int wave_sum[8];
    __shared__ int base_s;
    const int tid = threadIdx.x, lane = tid & 31, wave = tid >> 5;
    if (tid == 0) base_s = 0;
    __syncthreads();

    for (int t0 = 0; t0 < B_TOK; t0 += 256) {
        const int token = t0 + tid;
        int pred = 0; float w = 0.f;
#pragma unroll
        for (int k = 0; k < K_TOP; ++k) {
            if (topk_idx[token * K_TOP + k] == e) { pred = 1; w = topk_w[token * K_TOP + k]; }
        }
        const unsigned mask = __builtin_amdgcn_ballot_w32(pred != 0);
        const int prefix = __builtin_popcount(mask & ((1u << lane) - 1u));
        if (lane == 0) wave_sum[wave] = __builtin_popcount(mask);
        __syncthreads();
        int woff = 0, tot = 0;
#pragma unroll
        for (int i = 0; i < 8; ++i) { if (i < wave) woff += wave_sum[i]; tot += wave_sum[i]; }
        const int mybase = base_s;
        if (pred) {
            const int slot = mybase + woff + prefix;
            etok[e * B_TOK + slot] = token;
            ewt[e * B_TOK + slot]  = w;
        }
        __syncthreads();
        if (tid == 0) base_s = mybase + tot;
    }
    __syncthreads();
    if (tid == 0) counts[e] = base_s;
}

// ---------------------------------------------------------------------------
// Kernel 4: grouped expert GEMM, WMMA f16 -> f32 accumulate.
// grid = (6, 32, 16), block = 256 (8 waves as 4M x 2N).
// Block tile 128(M) x 128(N) x K64; wave tile 32x64 (8 WMMA accumulators).
// Double-buffered LDS; tiles fetched with global_load_async_to_lds_b128.
// WF16: expert weights pre-converted to f16 (fully async path).
// ---------------------------------------------------------------------------
#define LDK 72  // padded LDS row stride in halves (144B = 36 banks, conflict-free)

template <bool WF16>
__launch_bounds__(256)
__global__ void moe_grouped_gemm(const _Float16* __restrict__ xh,
                                 const _Float16* __restrict__ wh,
                                 const float*    __restrict__ wf,
                                 const float*    __restrict__ expert_b,
                                 const int*      __restrict__ counts,
                                 const int*      __restrict__ etok,
                                 const float*    __restrict__ ewt,
                                 float* __restrict__ out) {
    const int e  = blockIdx.z;
    const int mt = blockIdx.y;
    const int nt = blockIdx.x;
    const int cnt = counts[e];
    if (mt * 128 >= cnt) return;  // wave-uniform exit

    __shared__ __align__(16) _Float16 As[2][128][LDK];
    __shared__ __align__(16) _Float16 Bs[2][128][LDK];
    __shared__ int   Ts[128];
    __shared__ float Wt[128];

    const int tid = threadIdx.x;
    if (tid < 128) {
        const int row = mt * 128 + tid;
        if (row < cnt) { Ts[tid] = etok[e * B_TOK + row]; Wt[tid] = ewt[e * B_TOK + row]; }
        else           { Ts[tid] = 0;                     Wt[tid] = 0.f; }
    }
    __syncthreads();

    // staging maps: A tile 128x64 halves -> thread: row=tid>>1, 32 halves (4 b128)
    const int arow = tid >> 1;
    const int aseg = (tid & 1) * 32;
    const long abase = (long)Ts[arow] * S_DIM + aseg;
    // B tile 128x64 halves -> same split; B rows are P-columns
    const int brow = arow;
    const int bseg = aseg;
    const int ngl  = nt * 128 + brow;
    const int ngc  = (ngl < P_DIM) ? ngl : 0;  // clamp; garbage cols masked in epilogue
    const long bbase = ((long)e * P_DIM + ngc) * S_DIM + bseg;

    const int lane = tid & 31;
    const int wave = tid >> 5;
    const int wm   = wave & 3;   // row slab:  wm*32 (4 slabs -> 128 rows)
    const int wn   = wave >> 2;  // col slab:  wn*64 (2 slabs -> 128 cols)
    const int half = lane >> 4;
    const int l16  = lane & 15;
    const int ra0  = wm * 32 + l16, ra1 = ra0 + 16;
    const int nb0  = wn * 64 + l16;

    v8f c[2][4] = {};

    auto stageA = [&](int buf, int k0) {
#pragma unroll
        for (int j = 0; j < 4; ++j)
            async_ld_b128(xh + abase + k0 + j * 8, &As[buf][arow][aseg + j * 8]);
    };
    auto stageBh = [&](int buf, int k0) {
#pragma unroll
        for (int j = 0; j < 4; ++j)
            async_ld_b128(wh + bbase + k0 + j * 8, &Bs[buf][brow][bseg + j * 8]);
    };

    auto compute = [&](int buf) {
#pragma unroll
        for (int kc = 0; kc < 2; ++kc) {
            const int ko = kc * 32 + half * 8;  // lane's K-chunk (ISA 16-bit A layout)
            HF16 a[2], b[4];
            a[0].h[0] = *(const v8h*)&As[buf][ra0][ko];
            a[0].h[1] = *(const v8h*)&As[buf][ra0][ko + 16];
            a[1].h[0] = *(const v8h*)&As[buf][ra1][ko];
            a[1].h[1] = *(const v8h*)&As[buf][ra1][ko + 16];
#pragma unroll
            for (int ni = 0; ni < 4; ++ni) {
                b[ni].h[0] = *(const v8h*)&Bs[buf][nb0 + ni * 16][ko];
                b[ni].h[1] = *(const v8h*)&Bs[buf][nb0 + ni * 16][ko + 16];
            }
#pragma unroll
            for (int mi = 0; mi < 2; ++mi)
#pragma unroll
                for (int ni = 0; ni < 4; ++ni)
                    c[mi][ni] = __builtin_amdgcn_wmma_f32_16x16x32_f16(
                        false, a[mi].v, false, b[ni].v, (short)0, c[mi][ni], false, false);
        }
    };

    auto stageBf32_reg = [&](int k0, float4* pf) {
        const float* src = wf + bbase + k0;
#pragma unroll
        for (int j = 0; j < 8; ++j) pf[j] = *(const float4*)(src + j * 4);
    };
    auto storeBf32 = [&](int buf, const float4* pf) {
#pragma unroll
        for (int j = 0; j < 4; ++j) {
            const float4 f0 = pf[2 * j], f1 = pf[2 * j + 1];
            v8h hb;
            hb[0] = (_Float16)f0.x; hb[1] = (_Float16)f0.y; hb[2] = (_Float16)f0.z; hb[3] = (_Float16)f0.w;
            hb[4] = (_Float16)f1.x; hb[5] = (_Float16)f1.y; hb[6] = (_Float16)f1.z; hb[7] = (_Float16)f1.w;
            *(v8h*)&Bs[buf][brow][bseg + j * 8] = hb;
        }
    };

    // prologue: tile 0
    stageA(0, 0);
    if (WF16) {
        stageBh(0, 0);
    } else {
        float4 pf[8];
        stageBf32_reg(0, pf);
        storeBf32(0, pf);
    }
    async_wait0();
    __syncthreads();

    const int NSTEP = S_DIM / 64;  // 32
    for (int i = 0; i < NSTEP; ++i) {
        const int buf = i & 1;
        const bool more = (i + 1) < NSTEP;
        float4 pf[8];
        if (more) {
            stageA(buf ^ 1, (i + 1) * 64);  // async fire into other buffer
            if (WF16) stageBh(buf ^ 1, (i + 1) * 64);
            else      stageBf32_reg((i + 1) * 64, pf);  // loads stay pending under WMMAs
        }

        compute(buf);

        if (more) {
            if (!WF16) storeBf32(buf ^ 1, pf);  // convert + store after compute
            async_wait0();
        }
        __syncthreads();
    }

    // epilogue: C layout -> lane holds col l16, rows v + 8*half
    const int cbase = nt * 128 + wn * 64 + l16;
    float bias[4]; bool okc[4];
#pragma unroll
    for (int ni = 0; ni < 4; ++ni) {
        const int col = cbase + ni * 16;
        okc[ni]  = (col < P_DIM);
        bias[ni] = okc[ni] ? expert_b[e * P_DIM + col] : 0.f;
    }
#pragma unroll
    for (int mi = 0; mi < 2; ++mi) {
#pragma unroll
        for (int v = 0; v < 8; ++v) {
            const int lr = wm * 32 + mi * 16 + v + 8 * half;
            if ((mt * 128 + lr) < cnt) {
                const long orow = (long)Ts[lr] * P_DIM;
                const float w   = Wt[lr];
#pragma unroll
                for (int ni = 0; ni < 4; ++ni)
                    if (okc[ni])
                        atomicAdd(&out[orow + cbase + ni * 16], w * (c[mi][ni][v] + bias[ni]));
            }
        }
    }
}

// ---------------------------------------------------------------------------
extern "C" void kernel_launch(void* const* d_in, const int* in_sizes, int n_in,
                              void* d_out, int out_size, void* d_ws, size_t ws_size,
                              hipStream_t stream) {
    (void)in_sizes; (void)n_in;
    const float* x        = (const float*)d_in[0];
    const float* gate_w   = (const float*)d_in[1];
    const float* gate_b   = (const float*)d_in[2];
    const float* expert_w = (const float*)d_in[3];
    const float* expert_b = (const float*)d_in[4];
    float* out = (float*)d_out;

    // workspace layout
    char* ws = (char*)d_ws;
    size_t off = 0;
    _Float16* xh     = (_Float16*)(ws + off); off += (size_t)B_TOK * S_DIM * sizeof(_Float16);
    int*      tk_idx = (int*)     (ws + off); off += (size_t)B_TOK * K_TOP * sizeof(int);
    float*    tk_w   = (float*)   (ws + off); off += (size_t)B_TOK * K_TOP * sizeof(float);
    int*      counts = (int*)     (ws + off); off += 256;
    int*      etok   = (int*)     (ws + off); off += (size_t)E_NUM * B_TOK * sizeof(int);
    float*    ewt    = (float*)   (ws + off); off += (size_t)E_NUM * B_TOK * sizeof(float);
    _Float16* wh     = (_Float16*)(ws + off);
    const size_t need_big = off + (size_t)E_NUM * P_DIM * S_DIM * sizeof(_Float16);
    const bool big = (ws_size >= need_big);  // fixed per harness -> deterministic

    // output accumulated with atomics -> zero every call (graph-capture safe)
    hipMemsetAsync(d_out, 0, (size_t)out_size * sizeof(float), stream);

    convert_f32_to_f16<<<(B_TOK * S_DIM) / (256 * 8), 256, 0, stream>>>(x, xh);
    if (big)
        convert_f32_to_f16<<<(E_NUM * P_DIM * S_DIM) / (256 * 8), 256, 0, stream>>>(expert_w, wh);

    moe_gate_topk  <<<B_TOK / 8, 256, 0, stream>>>(x, gate_w, gate_b, tk_idx, tk_w);
    moe_build_groups<<<E_NUM, 256, 0, stream>>>(tk_idx, tk_w, etok, ewt, counts);

    dim3 grid((P_DIM + 127) / 128, B_TOK / 128, E_NUM);  // (6, 32, 16); blocks self-trim vs counts
    if (big)
        moe_grouped_gemm<true><<<grid, 256, 0, stream>>>(xh, wh, expert_w, expert_b,
                                                         counts, etok, ewt, out);
    else
        moe_grouped_gemm<false><<<grid, 256, 0, stream>>>(xh, wh, expert_w, expert_b,
                                                          counts, etok, ewt, out);
}
